// HRIU_7189775253851
// MI455X (gfx1250) — matrix operand order repaired and checked by reading, not compiled, not run on hardware
//
#include <hip/hip_runtime.h>

// ---------------- problem constants (match reference) ----------------
#define BB    2
#define HIMG  128
#define WIMG  128
#define CCH   256
#define NH    8
#define HD    32
#define WH    8
#define WW    8
#define PP    64
#define NQK   5
#define OBT   (NH*HD*NQK)     // 1280
#define NWH   (HIMG/WH)       // 16
#define NWW   (WIMG/WW)       // 16
#define NWIN  (BB*NWH*NWW)    // 512
#define SS    (1+HD+HD)       // 65
#define RDIM  225             // 15*15 relative offsets
#define RSTR  228             // padded row stride for M1/M2
#define EPSV  1e-5f

// ---------------- CDNA5 WMMA / TDM types ----------------
typedef __attribute__((ext_vector_type(16))) __bf16          v16bf;
typedef __attribute__((ext_vector_type(8)))  float           v8f;
typedef __attribute__((ext_vector_type(8)))  unsigned short  us8;
typedef __attribute__((ext_vector_type(4)))  unsigned int    uv4;
typedef __attribute__((ext_vector_type(8)))  int             iv8;
typedef __attribute__((ext_vector_type(4)))  int             iv4;

union BV {
  unsigned short u[16];
  us8            c[2];
  v16bf          v;
};

__device__ __forceinline__ unsigned short f2bf(float f) {
  // native convert -> v_cvt_*bf16_f32 (RNE), instead of manual bit-twiddle
  __bf16 h = (__bf16)f;
  return __builtin_bit_cast(unsigned short, h);
}
__device__ __forceinline__ float bf2f(unsigned short h) {
  union { unsigned u; float f; } cv; cv.u = ((unsigned)h) << 16;
  return cv.f;
}

__device__ __forceinline__ v8f wmma_bf16(const BV& a, const BV& b, v8f c) {
  // D(16x16 f32) = A(16x32 bf16) x B(32x16 bf16) + C
  return __builtin_amdgcn_wmma_f32_16x16x32_bf16(false, a.v, false, b.v,
                                                 (short)0, c, false, false);
}

// ---------------- LDS scratch (staging unions with attention scratch) ----
struct AttScratch {
  unsigned short m1[PP * RSTR];   // q_n . pat_n  (o,r)
  unsigned short m2[PP * RSTR];   // k_m . pat_m  (i,r)
  float          score[PP * 68];  // score (o,i), padded
};
union Scratch {
  float      stage[PP * CCH];     // TDM f32 landing buffer for x window
  AttScratch att;
};

// ---------------- fused kernel: one workgroup per 8x8 window ----------------
__global__ __launch_bounds__(256)
void HRIU_swin_fused(const float* __restrict__ xg,
                     const float* __restrict__ Wp,    // (256,1280)
                     const float* __restrict__ bprep, // (1280)
                     const float* __restrict__ intr,  // (8,15,15,65)
                     const float* __restrict__ Wf,    // (256,256)
                     const float* __restrict__ bfin,  // (256)
                     float* __restrict__ outg)        // (2,128,128,256)
{
  // LDS total ~158 KB: two workgroups fit per 320 KB WGP
  __shared__ __align__(16) Scratch        sU;
  __shared__ __align__(16) unsigned short sXw[PP * CCH];      // x window, bf16
  __shared__ __align__(16) unsigned short sT[NQK * PP * HD];  // per-head q_n,q,k,k_m,v
  __shared__ __align__(16) unsigned short sOut[PP * CCH];     // attention out, bf16

  const int tid  = threadIdx.x;
  const int wave = tid >> 5;
  const int lane = tid & 31;
  const int hiL  = lane >> 4;     // WMMA lane half
  const int ln   = lane & 15;

  const int win  = blockIdx.x;          // (b, wi, wj)
  const int bIdx = win >> 8;
  const int wi   = (win >> 4) & 15;
  const int wj   = win & 15;
  const int rowBase = (bIdx * HIMG + wi * WH) * WIMG + wj * WW; // flat pixel base

#if __has_builtin(__builtin_amdgcn_tensor_load_to_lds) && __has_builtin(__builtin_amdgcn_s_wait_tensorcnt)
  // ---- Phase 0a: TDM — DMA the window tile (8 lines x 2048 f32) into LDS ----
  if (wave == 0) {
    const unsigned long long ga =
        (unsigned long long)(uintptr_t)(xg + (size_t)rowBase * CCH);
    const unsigned ldsA = (unsigned)(uintptr_t)(&sU.stage[0]);

    uv4 g0;                                   // D# group 0 (ISA 8.3)
    g0.x = 1u;                                // count=1 (valid), user mode
    g0.y = ldsA;                              // lds_addr
    g0.z = (unsigned)(ga & 0xFFFFFFFFu);      // global_addr[31:0]
    g0.w = (unsigned)((ga >> 32) & 0x01FFFFFFu) | (2u << 30); // [56:32] | type=2

    const unsigned td0 = 2048u, td1 = 8u;     // tensor dims (elements)
    const unsigned tl0 = 2048u, tl1 = 8u;     // tile dims
    const unsigned st0 = (unsigned)(WIMG * CCH);  // 32768 elements / line
    iv8 g1;                                   // D# group 1 (ISA 8.4)
    g1[0] = (int)(2u << 16);                  // data_size=2 (4 bytes)
    g1[1] = (int)((td0 & 0xFFFFu) << 16);     // tensor_dim0[15:0]
    g1[2] = (int)((td0 >> 16) | ((td1 & 0xFFFFu) << 16));
    g1[3] = (int)((td1 >> 16) | (tl0 << 16)); // tile_dim0
    g1[4] = (int)(tl1 | (0u << 16));          // tile_dim1, tile_dim2=0
    g1[5] = (int)st0;                         // tensor_dim0_stride[31:0]
    g1[6] = 0;                                // stride0[47:32] | stride1[15:0]
    g1[7] = 0;                                // stride1[47:16]

    iv4 gz4 = {0, 0, 0, 0};                   // groups 2/3 unused (2D tensor)
    iv8 gz8 = {0, 0, 0, 0, 0, 0, 0, 0};       // extra group slot (6-arg variant)
    __builtin_amdgcn_tensor_load_to_lds(g0, g1, gz4, gz4, gz8, 0);
    __builtin_amdgcn_s_wait_tensorcnt(0);
  }
  __syncthreads();

  // ---- Phase 0b: convert staged f32 -> bf16 operand buffer ----
  for (int idx = tid; idx < PP * CCH; idx += 256)
    sXw[idx] = f2bf(sU.stage[idx]);
  __syncthreads();
#else
  // ---- Fallback: direct vector loads + convert ----
  for (int it = 0; it < 16; ++it) {
    int gi = (tid + it * 256) * 4;
    int p  = gi >> 8;
    int c  = gi & 255;
    int pixel = rowBase + (p >> 3) * WIMG + (p & 7);
    const float4 v4 = *reinterpret_cast<const float4*>(xg + (size_t)pixel * CCH + c);
    sXw[p * CCH + c + 0] = f2bf(v4.x);
    sXw[p * CCH + c + 1] = f2bf(v4.y);
    sXw[p * CCH + c + 2] = f2bf(v4.z);
    sXw[p * CCH + c + 3] = f2bf(v4.w);
  }
  __syncthreads();
#endif

  // ---- per-head pipeline (heads sequential; all 8 waves cooperate) ----
  for (int h = 0; h < NH; ++h) {
    // Phase 1: GEMM1 slice for head h: (64x256)@(256x160) -> sT[qk][p][d]
    for (int k5 = 0; k5 < 5; ++k5) {
      const int tt = wave + (k5 << 3);   // 0..39
      const int mt = tt & 3;
      const int nt = tt >> 2;            // 0..9
      const int qk = nt >> 1;
      const int dh = (nt & 1) << 4;
      const int col = qk * (NH * HD) + h * HD + dh + ln;
      v8f acc = {};
      for (int s = 0; s < 8; ++s) {
        BV a, b;
        const int p = mt * 16 + ln;
        const unsigned short* ap = &sXw[p * CCH + s * 32 + hiL * 16];
        a.c[0] = *(const us8*)ap;
        a.c[1] = *(const us8*)(ap + 8);
        const int kb = s * 32 + hiL * 16;
        #pragma unroll
        for (int j = 0; j < 16; ++j)
          b.u[j] = f2bf(Wp[(size_t)(kb + j) * OBT + col]);
        acc = wmma_bf16(a, b, acc);
      }
      const float bias = bprep[col];
      #pragma unroll
      for (int r = 0; r < 8; ++r) {
        const int p = mt * 16 + hiL * 8 + r;
        sT[(qk * PP + p) * HD + dh + ln] = f2bf(acc[r] + bias);
      }
    }
    __syncthreads();

    // Phase 2: M1[o,r]=q_n.intr[...,1:33], M2[i,r]=k_m.intr[...,33:65]
    for (int k15 = 0; k15 < 15; ++k15) {
      const int tt   = wave + (k15 << 3);    // 0..119
      const int isM2 = (tt >= 60) ? 1 : 0;
      const int u    = tt - (isM2 ? 60 : 0);
      const int mt   = u & 3;
      const int nt   = u >> 2;               // 0..14
      const int qk   = isM2 ? 3 : 0;
      const int soff = isM2 ? (1 + HD) : 1;
      unsigned short* dst = isM2 ? sU.att.m2 : sU.att.m1;
      BV a, b;
      {
        const int p = mt * 16 + ln;
        const unsigned short* ap = &sT[(qk * PP + p) * HD + hiL * 16];
        a.c[0] = *(const us8*)ap;
        a.c[1] = *(const us8*)(ap + 8);
      }
      const int rr = nt * 16 + ln;
      if (rr < RDIM) {
        const float* src = intr + (((size_t)h * 15 + (rr / 15)) * 15 + (rr % 15)) * SS
                         + soff + hiL * 16;
        #pragma unroll
        for (int j = 0; j < 16; ++j) b.u[j] = f2bf(src[j]);
      } else {
        #pragma unroll
        for (int j = 0; j < 16; ++j) b.u[j] = 0;
      }
      v8f acc = {};
      acc = wmma_bf16(a, b, acc);
      #pragma unroll
      for (int r = 0; r < 8; ++r) {
        const int m = mt * 16 + hiL * 8 + r;
        if (rr < RDIM) dst[m * RSTR + rr] = f2bf(acc[r]);
      }
    }
    __syncthreads();

    // Phase 3: score = QK^T + bias + M1-gather + M2-gather. 16 tiles, 2/wave.
    for (int k2 = 0; k2 < 2; ++k2) {
      const int tt = wave + (k2 << 3);   // 0..15
      const int mt = tt & 3;
      const int it = tt >> 2;
      BV a, b;
      {
        const int p = mt * 16 + ln;                       // A = q  (qk=1)
        const unsigned short* ap = &sT[(1 * PP + p) * HD + hiL * 16];
        a.c[0] = *(const us8*)ap;
        a.c[1] = *(const us8*)(ap + 8);
      }
      {
        const int i = it * 16 + ln;                       // B = k^T (qk=2)
        const unsigned short* bp2 = &sT[(2 * PP + i) * HD + hiL * 16];
        b.c[0] = *(const us8*)bp2;
        b.c[1] = *(const us8*)(bp2 + 8);
      }
      v8f acc = {};
      acc = wmma_bf16(a, b, acc);
      #pragma unroll
      for (int r = 0; r < 8; ++r) {
        const int m  = mt * 16 + hiL * 8 + r;
        const int i  = it * 16 + ln;
        const int oy = m >> 3, ox = m & 7, iy = i >> 3, ix = i & 7;
        const int ry = oy - iy + 7, rx = ox - ix + 7;
        const float bv = intr[(((size_t)h * 15 + ry) * 15 + rx) * SS];
        const float t1 = bf2f(sU.att.m1[m * RSTR + ry * 15 + rx]);
        const float t2 = bf2f(sU.att.m2[i * RSTR + ry * 15 + rx]);
        sU.att.score[m * 68 + i] = acc[r] + bv + t1 + t2;
      }
    }
    __syncthreads();

    // Phase 4: per-row normalization over i (mean, ddof=1 var). 4 thr/row.
    {
      const int row = tid >> 2;
      const int sub = tid & 3;
      float sum = 0.f, sq = 0.f;
      #pragma unroll
      for (int j = 0; j < 16; ++j) {
        const float v = sU.att.score[row * 68 + sub * 16 + j];
        sum += v; sq += v * v;
      }
      sum += __shfl_xor(sum, 1, 32);  sq += __shfl_xor(sq, 1, 32);
      sum += __shfl_xor(sum, 2, 32);  sq += __shfl_xor(sq, 2, 32);
      const float mean = sum * (1.f / 64.f);
      const float var  = (sq - 64.f * mean * mean) * (1.f / 63.f);
      const float rstd = rsqrtf(var + EPSV);
      #pragma unroll
      for (int j = 0; j < 16; ++j) {
        const float v = sU.att.score[row * 68 + sub * 16 + j];
        sU.att.score[row * 68 + sub * 16 + j] = (v - mean) * rstd;
      }
    }
    __syncthreads();

    // Phase 5: out_h = score @ V  (64x64 @ 64x32). 8 tiles, 1/wave, K=2 slices.
    {
      const int mt = wave & 3;
      const int nt = wave >> 2;
      v8f acc = {};
      for (int s = 0; s < 2; ++s) {
        BV a, b;
        const int m = mt * 16 + ln;
        const float* ap = &sU.att.score[m * 68 + s * 32 + hiL * 16];
        #pragma unroll
        for (int j = 0; j < 8; ++j) {
          a.u[j]     = f2bf(ap[j]);
          a.u[8 + j] = f2bf(ap[8 + j]);
        }
        const int d = nt * 16 + ln;
        #pragma unroll
        for (int j = 0; j < 16; ++j) {
          const int i = s * 32 + hiL * 16 + j;              // B = V (qk=4)
          b.u[j] = sT[(4 * PP + i) * HD + d];
        }
        acc = wmma_bf16(a, b, acc);
      }
      #pragma unroll
      for (int r = 0; r < 8; ++r) {
        const int m = mt * 16 + hiL * 8 + r;
        sOut[m * CCH + h * HD + nt * 16 + ln] = f2bf(acc[r]);
      }
    }
    __syncthreads();   // also protects sT reuse by next head
  }

  // ---- Phase 6: final projection (64x256)@(256x256)+b_fin -> d_out ----
  for (int k8 = 0; k8 < 8; ++k8) {
    const int tt = wave + (k8 << 3);    // 0..63
    const int mt = tt & 3;
    const int nt = tt >> 2;             // 0..15
    const int n  = nt * 16 + ln;
    v8f acc = {};
    for (int s = 0; s < 8; ++s) {
      BV a, b;
      const int p = mt * 16 + ln;
      const unsigned short* ap = &sOut[p * CCH + s * 32 + hiL * 16];
      a.c[0] = *(const us8*)ap;
      a.c[1] = *(const us8*)(ap + 8);
      const int kb = s * 32 + hiL * 16;
      #pragma unroll
      for (int j = 0; j < 16; ++j)
        b.u[j] = f2bf(Wf[(size_t)(kb + j) * CCH + n]);
      acc = wmma_bf16(a, b, acc);
    }
    const float bias = bfin[n];
    #pragma unroll
    for (int r = 0; r < 8; ++r) {
      const int m = mt * 16 + hiL * 8 + r;
      const int pixel = rowBase + (m >> 3) * WIMG + (m & 7);
      outg[(size_t)pixel * CCH + n] = acc[r] + bias;
    }
  }
}

extern "C" void kernel_launch(void* const* d_in, const int* in_sizes, int n_in,
                              void* d_out, int out_size, void* d_ws, size_t ws_size,
                              hipStream_t stream) {
  (void)in_sizes; (void)n_in; (void)d_ws; (void)ws_size; (void)out_size;
  const float* x      = (const float*)d_in[0];
  const float* W_prep = (const float*)d_in[1];
  const float* b_prep = (const float*)d_in[2];
  const float* intrin = (const float*)d_in[3];
  const float* W_fin  = (const float*)d_in[4];
  const float* b_fin  = (const float*)d_in[5];
  float* out = (float*)d_out;
  HRIU_swin_fused<<<dim3(NWIN), dim3(256), 0, stream>>>(
      x, W_prep, b_prep, intrin, W_fin, b_fin, out);
}
